// TransformerLayer_46170898432689
// MI455X (gfx1250) — matrix operand order repaired and checked by reading, not compiled, not run on hardware
//
#include <hip/hip_runtime.h>
#include <hip/hip_fp16.h>

// ---------------------------------------------------------------------------
// TransformerLayer for MI455X (gfx1250): f16 WMMA GEMMs + fused softmax.
// Dead-code-eliminates Wv / attended-values (unused in reference output).
// ---------------------------------------------------------------------------

typedef __attribute__((ext_vector_type(16))) _Float16 v16h;
typedef __attribute__((ext_vector_type(8)))  float    v8f;

union HFrag { v16h v; uint4 q[2]; };

#define D_MODEL 2048
#define SEQ     1024
#define NBATCH  4
#define NHEAD   32
#define HDIM    64

// ---------------------------------------------------------------------------
// fp32 -> f16 elementwise (weight conversion)
// ---------------------------------------------------------------------------
__global__ __launch_bounds__(256) void cvt_f32_f16(const float* __restrict__ s,
                                                   _Float16* __restrict__ d, int n) {
  int i = blockIdx.x * 256 + threadIdx.x;
  if (i < n) d[i] = (_Float16)s[i];
}

// ---------------------------------------------------------------------------
// Per-row: h = LN1(x); x2 = h + x; h2 = LN2(x2).  One 256-thread block per row.
// ---------------------------------------------------------------------------
__global__ __launch_bounds__(256) void ln_dual(
    const float* __restrict__ x,
    const float* __restrict__ s1, const float* __restrict__ b1,
    const float* __restrict__ s2, const float* __restrict__ b2,
    _Float16* __restrict__ h16, float* __restrict__ x2out,
    _Float16* __restrict__ h2_16) {
  __shared__ float redS[8], redQ[8];
  const int tid = threadIdx.x, lane = tid & 31, wave = tid >> 5;
  const size_t base = (size_t)blockIdx.x * D_MODEL;

  float xv[8];
#pragma unroll
  for (int j = 0; j < 8; ++j) xv[j] = x[base + tid + 256 * j];

  float s = 0.f, qq = 0.f;
#pragma unroll
  for (int j = 0; j < 8; ++j) { s += xv[j]; qq += xv[j] * xv[j]; }
  for (int off = 16; off > 0; off >>= 1) {
    s  += __shfl_xor(s, off, 32);
    qq += __shfl_xor(qq, off, 32);
  }
  if (lane == 0) { redS[wave] = s; redQ[wave] = qq; }
  __syncthreads();
  float S = 0.f, Q = 0.f;
#pragma unroll
  for (int w = 0; w < 8; ++w) { S += redS[w]; Q += redQ[w]; }
  float m  = S * (1.0f / D_MODEL);
  float vr = Q * (1.0f / D_MODEL) - m * m;
  float rs = rsqrtf(vr + 1e-5f);

  float x2v[8];
#pragma unroll
  for (int j = 0; j < 8; ++j) {
    int c = tid + 256 * j;
    float hh = (xv[j] - m) * rs * s1[c] + b1[c];
    h16[base + c] = (_Float16)hh;
    x2v[j] = hh + xv[j];
    x2out[base + c] = x2v[j];
  }

  s = 0.f; qq = 0.f;
#pragma unroll
  for (int j = 0; j < 8; ++j) { s += x2v[j]; qq += x2v[j] * x2v[j]; }
  for (int off = 16; off > 0; off >>= 1) {
    s  += __shfl_xor(s, off, 32);
    qq += __shfl_xor(qq, off, 32);
  }
  __syncthreads();  // LDS reuse
  if (lane == 0) { redS[wave] = s; redQ[wave] = qq; }
  __syncthreads();
  S = 0.f; Q = 0.f;
#pragma unroll
  for (int w = 0; w < 8; ++w) { S += redS[w]; Q += redQ[w]; }
  m  = S * (1.0f / D_MODEL);
  vr = Q * (1.0f / D_MODEL) - m * m;
  rs = rsqrtf(vr + 1e-5f);
#pragma unroll
  for (int j = 0; j < 8; ++j) {
    int c = tid + 256 * j;
    float hh = (x2v[j] - m) * rs * s2[c] + b2[c];
    h2_16[base + c] = (_Float16)hh;
  }
}

// ---------------------------------------------------------------------------
// f16 WMMA GEMM: C[M,N] = A[M,K] x B[K,N]  (A,B f16 row-major, fp32 accum).
// Block tile 64x256 (8 waves as 2x4), wave tile 32x64, K-step 32:
// 8 WMMAs per wave per K-step from 12 aligned b128 LDS fragment reads.
// LDS: A row-major [64][32]; B stored K-contiguous per column ([256][40-half
// padded rows]) so B-fragments use the same contiguous K order as A (K 0-15/16-31
// per half-wave) as two aligned b128 reads. B staging packs two K-rows per
// dword -> 16 ds_write_b32 per thread instead of 32 ds_write_b16.
// Next K-tile is prefetched with global_prefetch_b8.
// ---------------------------------------------------------------------------
template <bool BIAS, bool RELU, bool RES, bool OUT16>
__global__ __launch_bounds__(256) void gemm_wmma(
    const _Float16* __restrict__ A, const _Float16* __restrict__ Bw,
    const float* __restrict__ bias, const float* __restrict__ res,
    _Float16* __restrict__ o16, float* __restrict__ o32,
    int M, int N, int K) {
  __shared__ __attribute__((aligned(16))) _Float16 lsA[64][32];
  __shared__ __attribute__((aligned(16))) _Float16 lsB[256][40];

  const int tid = threadIdx.x;
  const int lane = tid & 31, wave = tid >> 5;
  const int wm = wave >> 2, wn = wave & 3;          // 2 (M) x 4 (N) waves
  const int m0 = blockIdx.y * 64, n0 = blockIdx.x * 256;

  v8f acc[2][4] = {};

  const int ar  = tid >> 2, ac = (tid & 3) * 8;     // A stage: 16B per thread
  const int bkp = tid >> 4;                         // B stage: k-pair 0..15
  const int bnb = (tid & 15) * 16;                  //          16-col chunk

  for (int k0 = 0; k0 < K; k0 += 32) {
    __syncthreads();
    *(uint4*)&lsA[ar][ac] = *(const uint4*)&A[(size_t)(m0 + ar) * K + k0 + ac];
    {
      const _Float16* b0p = &Bw[(size_t)(k0 + 2 * bkp) * N + n0 + bnb];
      const _Float16* b1p = b0p + N;                // next K row
      uint4 r0a = *(const uint4*)(b0p);
      uint4 r0b = *(const uint4*)(b0p + 8);
      uint4 r1a = *(const uint4*)(b1p);
      uint4 r1b = *(const uint4*)(b1p + 8);
      const unsigned short* h0 = (const unsigned short*)&r0a;
      const unsigned short* h1 = (const unsigned short*)&r1a;
#pragma unroll
      for (int j = 0; j < 8; ++j)
        *(unsigned*)&lsB[bnb + j][2 * bkp] =
            (unsigned)h0[j] | ((unsigned)h1[j] << 16);
      h0 = (const unsigned short*)&r0b;
      h1 = (const unsigned short*)&r1b;
#pragma unroll
      for (int j = 0; j < 8; ++j)
        *(unsigned*)&lsB[bnb + 8 + j][2 * bkp] =
            (unsigned)h0[j] | ((unsigned)h1[j] << 16);
    }
    if (k0 + 32 < K) {  // global_prefetch_b8 of next K tile
      __builtin_prefetch(&A[(size_t)(m0 + ar) * K + k0 + 32 + ac], 0, 1);
      __builtin_prefetch(&Bw[(size_t)(k0 + 32 + 2 * bkp) * N + n0 + bnb], 0, 1);
    }
    __syncthreads();

    HFrag af[2];
    const int akb = (lane < 16) ? 0 : 16;
#pragma unroll
    for (int mt = 0; mt < 2; ++mt) {
      const int am = wm * 32 + mt * 16 + (lane & 15);
      af[mt].q[0] = *(const uint4*)&lsA[am][akb];
      af[mt].q[1] = *(const uint4*)&lsA[am][akb + 8];
    }
    const int bkb = (lane < 16) ? 0 : 16;
#pragma unroll
    for (int nt = 0; nt < 4; ++nt) {
      HFrag bf;
      const int nl = wn * 64 + nt * 16 + (lane & 15);
      bf.q[0] = *(const uint4*)&lsB[nl][bkb];
      bf.q[1] = *(const uint4*)&lsB[nl][bkb + 8];
#pragma unroll
      for (int mt = 0; mt < 2; ++mt)
        acc[mt][nt] = __builtin_amdgcn_wmma_f32_16x16x32_f16(
            false, af[mt].v, false, bf.v, (short)0, acc[mt][nt], false, false);
    }
  }

#pragma unroll
  for (int nt = 0; nt < 4; ++nt) {
    const int col = n0 + wn * 64 + nt * 16 + (lane & 15);
    const float bv = BIAS ? bias[col] : 0.0f;
#pragma unroll
    for (int mt = 0; mt < 2; ++mt) {
      const int rbase = m0 + wm * 32 + mt * 16 + ((lane >= 16) ? 8 : 0);
#pragma unroll
      for (int r = 0; r < 8; ++r) {
        const int row = rbase + r;
        float v = acc[mt][nt][r];
        if (BIAS) v += bv;
        if (RELU) v = fmaxf(v, 0.0f);
        if (RES)  v += res[(size_t)row * N + col];
        if (OUT16) o16[(size_t)row * N + col] = (_Float16)v;
        else       o32[(size_t)row * N + col] = v;
      }
    }
  }
}

// ---------------------------------------------------------------------------
// Fused scores + softmax: one block per (b, head, 16 query rows).
// Q/K fragments load straight from global (head slice is K-contiguous, so
// each fragment = two aligned 16B loads). 8 waves x 8 col-tiles x 2 WMMAs
// fill a 16x1024 fp32 score strip in LDS (64KB); each wave softmaxes 2 rows.
// ---------------------------------------------------------------------------
__global__ __launch_bounds__(256) void attn_softmax(
    const _Float16* __restrict__ q, const _Float16* __restrict__ k,
    float* __restrict__ probs) {
  __shared__ float sc[16][SEQ];  // 64 KB
  const int tid = threadIdx.x, lane = tid & 31, wave = tid >> 5;
  const int i0 = blockIdx.x * 16;
  const int h  = blockIdx.y;
  const int b  = blockIdx.z;

  const int arow = i0 + (lane & 15);
  const int akb  = (lane < 16) ? 0 : 16;
  const _Float16* qp = q + (size_t)(b * SEQ + arow) * D_MODEL + h * HDIM;
  HFrag a0, a1;
  a0.q[0] = *(const uint4*)(qp + akb);
  a0.q[1] = *(const uint4*)(qp + akb + 8);
  a1.q[0] = *(const uint4*)(qp + 32 + akb);
  a1.q[1] = *(const uint4*)(qp + 32 + akb + 8);

  const int bkb = (lane < 16) ? 0 : 16;
  const int rlo = (lane >= 16) ? 8 : 0;

  for (int t = 0; t < 8; ++t) {
    const int j0   = (wave + 8 * t) * 16;
    const int krow = j0 + (lane & 15);
    const _Float16* kp = k + (size_t)(b * SEQ + krow) * D_MODEL + h * HDIM;
    HFrag b0, b1;
    b0.q[0] = *(const uint4*)(kp + bkb);
    b0.q[1] = *(const uint4*)(kp + bkb + 8);
    b1.q[0] = *(const uint4*)(kp + 32 + bkb);
    b1.q[1] = *(const uint4*)(kp + 32 + bkb + 8);
    v8f acc = {};
    acc = __builtin_amdgcn_wmma_f32_16x16x32_f16(false, a0.v, false, b0.v,
                                                 (short)0, acc, false, false);
    acc = __builtin_amdgcn_wmma_f32_16x16x32_f16(false, a1.v, false, b1.v,
                                                 (short)0, acc, false, false);
#pragma unroll
    for (int r = 0; r < 8; ++r) sc[rlo + r][j0 + (lane & 15)] = acc[r];
  }
  __syncthreads();

  const float scale = 0.022097086912079608f;  // 1/sqrt(D), per reference
#pragma unroll
  for (int rr = 0; rr < 2; ++rr) {
    const int row = wave * 2 + rr;
    float mx = -3.4e38f;
    for (int t = 0; t < 32; ++t) mx = fmaxf(mx, sc[row][lane + 32 * t]);
    for (int off = 16; off > 0; off >>= 1) mx = fmaxf(mx, __shfl_xor(mx, off, 32));
    float sum = 0.f;
    for (int t = 0; t < 32; ++t) {
      float e = __expf((sc[row][lane + 32 * t] - mx) * scale);
      sc[row][lane + 32 * t] = e;
      sum += e;
    }
    for (int off = 16; off > 0; off >>= 1) sum += __shfl_xor(sum, off, 32);
    const float inv = 1.0f / sum;
    float* op = probs + ((size_t)(b * NHEAD + h) * SEQ + i0 + row) * SEQ;
    for (int t = 0; t < 32; ++t) op[lane + 32 * t] = sc[row][lane + 32 * t] * inv;
  }
}

// ---------------------------------------------------------------------------
extern "C" void kernel_launch(void* const* d_in, const int* in_sizes, int n_in,
                              void* d_out, int out_size, void* d_ws, size_t ws_size,
                              hipStream_t stream) {
  (void)in_sizes; (void)n_in; (void)out_size; (void)ws_size;

  const float* x   = (const float*)d_in[0];
  const float* Wq  = (const float*)d_in[1];
  const float* Wk  = (const float*)d_in[2];
  // d_in[3] (Wv): dead in the reference — attended values are never used.
  const float* l1s = (const float*)d_in[4];
  const float* l1b = (const float*)d_in[5];
  const float* l2s = (const float*)d_in[6];
  const float* l2b = (const float*)d_in[7];
  const float* f1w = (const float*)d_in[8];
  const float* f1b = (const float*)d_in[9];
  const float* f2w = (const float*)d_in[10];
  const float* f2b = (const float*)d_in[11];

  float* out   = (float*)d_out;                        // [4,1024,2048]
  float* probs = out + (size_t)NBATCH * SEQ * D_MODEL; // [4,32,1024,1024]

  char* w = (char*)d_ws;
  const size_t actB = (size_t)NBATCH * SEQ * D_MODEL;   // 8,388,608 elements
  const size_t wgtB = (size_t)D_MODEL * D_MODEL;        // 4,194,304 elements
  _Float16* h16   = (_Float16*)w; w += actB * 2;
  _Float16* h2    = (_Float16*)w; w += actB * 2;
  float*    x2    = (float*)w;    w += actB * 4;
  _Float16* q16   = (_Float16*)w; w += actB * 2;
  _Float16* k16   = (_Float16*)w; w += actB * 2;
  _Float16* y1    = (_Float16*)w; w += actB * 2;
  _Float16* Wq16  = (_Float16*)w; w += wgtB * 2;
  _Float16* Wk16  = (_Float16*)w; w += wgtB * 2;
  _Float16* F1w16 = (_Float16*)w; w += wgtB * 2;
  _Float16* F2w16 = (_Float16*)w; w += wgtB * 2;   // total ~151 MB scratch

  const int wn = (int)wgtB;
  cvt_f32_f16<<<wn / 256, 256, 0, stream>>>(Wq,  Wq16,  wn);
  cvt_f32_f16<<<wn / 256, 256, 0, stream>>>(Wk,  Wk16,  wn);
  cvt_f32_f16<<<wn / 256, 256, 0, stream>>>(f1w, F1w16, wn);
  cvt_f32_f16<<<wn / 256, 256, 0, stream>>>(f2w, F2w16, wn);

  ln_dual<<<NBATCH * SEQ, 256, 0, stream>>>(x, l1s, l1b, l2s, l2b, h16, x2, h2);

  dim3 ggrid(D_MODEL / 256, (NBATCH * SEQ) / 64);
  // q = h @ Wq ; k = h @ Wk  (f16 out for attention)
  gemm_wmma<false, false, false, true><<<ggrid, 256, 0, stream>>>(
      h16, Wq16, nullptr, nullptr, q16, nullptr, NBATCH * SEQ, D_MODEL, D_MODEL);
  gemm_wmma<false, false, false, true><<<ggrid, 256, 0, stream>>>(
      h16, Wk16, nullptr, nullptr, k16, nullptr, NBATCH * SEQ, D_MODEL, D_MODEL);

  attn_softmax<<<dim3(SEQ / 16, NHEAD, NBATCH), 256, 0, stream>>>(q16, k16, probs);

  // y1 = relu(h2 @ ff1_w + ff1_b)  (f16 out)
  gemm_wmma<true, true, false, true><<<ggrid, 256, 0, stream>>>(
      h2, F1w16, f1b, nullptr, y1, nullptr, NBATCH * SEQ, D_MODEL, D_MODEL);
  // out = y1 @ ff2_w + ff2_b + x2  (fp32 out, fused residual)
  gemm_wmma<true, false, true, false><<<ggrid, 256, 0, stream>>>(
      y1, F2w16, f2b, x2, nullptr, out, NBATCH * SEQ, D_MODEL, D_MODEL);
}
